// DualEncoderLSTM_53077205844744
// MI455X (gfx1250) — matrix operand order repaired and checked by reading, not compiled, not run on hardware
//
#include <hip/hip_runtime.h>
#include <hip/hip_bf16.h>

// MI455X (gfx1250) dual-encoder peephole LSTM.
// Batched recurrence (query rows 0..255, reply rows 256..511): per step one
// fused kernel does embedding-gather + [512x768]@[768x2048] bf16 WMMA GEMM
// (f32 accumulate, ping-pong software-pipelined fragment loads), then an f32
// gate kernel. Weights bf16-repacked once into WMMA-B fragment layout (L2
// resident). echo encoder == query encoder, so it is reused.

typedef __bf16 bf16_t;
typedef __attribute__((ext_vector_type(16))) __bf16 v16bf;
typedef __attribute__((ext_vector_type(8)))  __bf16 v8bf;
typedef __attribute__((ext_vector_type(8)))  float  v8f;

#define BATCH 256
#define SEQT  160
#define EDIM  256
#define HDIM  512
#define GDIM  2048   // 4*H
#define KDIM  768    // E+H
#define ROWS  512    // 2*B

// ---------------------------------------------------------------------------
// Repack row-major f32 [K x N] into WMMA-B fragment-contiguous bf16:
//   dst[((nt*kTiles+kt)*32 + lane)*16 + j] =
//       src[(kt*32 + (lane/16)*8 + (j%8) + (j/8)*16)*N + nt*16 + lane%16]
// (dense 16x16x32 B layout: lane holds column n=lane%16, k=(lane/16)*8+(j%8)+(j/8)*16)
// ---------------------------------------------------------------------------
__global__ void repack_w(const float* __restrict__ src, bf16_t* __restrict__ dst,
                         int K, int N) {
  const int kTiles = K >> 5;
  long id = (long)blockIdx.x * blockDim.x + threadIdx.x;
  if (id >= (long)K * N) return;
  int  j  = (int)(id & 15);
  int  l  = (int)((id >> 4) & 31);
  long f  = id >> 9;                 // fragment index = nt*kTiles + kt
  int  kt = (int)(f % kTiles);
  int  nt = (int)(f / kTiles);
  int  k  = (kt << 5) + ((l >> 4) << 3) + (j & 7) + ((j >> 3) << 4);
  int  n  = (nt << 4) + (l & 15);
  dst[id] = (bf16_t)src[(long)k * N + n];
}

__global__ void init_state(float* __restrict__ cst, bf16_t* __restrict__ hbuf) {
  int idx = blockIdx.x * blockDim.x + threadIdx.x;   // ROWS*HDIM
  cst[idx]  = 0.0f;
  hbuf[idx] = (bf16_t)0.0f;
}

// A fragment (16x32 bf16): lane holds row m=lane%16; two contiguous 8-elem
// runs at k0+(lane/16)*8 and k0+(lane/16)*8+16.
__device__ __forceinline__ v16bf load_afrag(const bf16_t* p) {
  v8bf lo = *(const v8bf*)(p);
  v8bf hi = *(const v8bf*)(p + 16);
  v16bf a;
#pragma unroll
  for (int j = 0; j < 8; ++j) { a[j] = lo[j]; a[j + 8] = hi[j]; }
  return a;
}

// Same fragment but gathered from an f32 embedding row, converted to bf16.
__device__ __forceinline__ v16bf load_emb_frag(const float* p) {
  float4 x0 = *(const float4*)(p);
  float4 x1 = *(const float4*)(p + 4);
  float4 x2 = *(const float4*)(p + 16);
  float4 x3 = *(const float4*)(p + 20);
  v16bf a;
  a[0]  = (bf16_t)x0.x; a[1]  = (bf16_t)x0.y; a[2]  = (bf16_t)x0.z; a[3]  = (bf16_t)x0.w;
  a[4]  = (bf16_t)x1.x; a[5]  = (bf16_t)x1.y; a[6]  = (bf16_t)x1.z; a[7]  = (bf16_t)x1.w;
  a[8]  = (bf16_t)x2.x; a[9]  = (bf16_t)x2.y; a[10] = (bf16_t)x2.z; a[11] = (bf16_t)x2.w;
  a[12] = (bf16_t)x3.x; a[13] = (bf16_t)x3.y; a[14] = (bf16_t)x3.z; a[15] = (bf16_t)x3.w;
  return a;
}

// Fused gather + GEMM: Z = [emb(tok_t) | H] @ Bpacked.
// K-tiles [0, embKt) come from the f32 embedding table (on-the-fly bf16
// convert); K-tiles [embKt, kTiles) come from the bf16 H state (lda = HDIM).
// Block: 8 waves (2x4), block tile 64x256, wave tile 32x64 (8 WMMA/K-iter).
// Two named fragment sets ping-pong across a 2x-unrolled K loop: loads for
// kt+2/kt+3 fly under the WMMAs for kt/kt+1, with no register copies.
__global__ __launch_bounds__(256)
void gemm_fused(const int* __restrict__ qtok, const int* __restrict__ rtok,
                const float* __restrict__ emb, int t,
                const bf16_t* __restrict__ H,
                const bf16_t* __restrict__ Bp,
                float* __restrict__ Z, int ldc, int kTiles, int embKt) {
  const int lane  = threadIdx.x & 31;
  const int wave  = threadIdx.x >> 5;
  const int m0    = blockIdx.x * 64 + (wave >> 2) * 32;
  const int n0    = blockIdx.y * 256 + (wave & 3) * 64;
  const int lhalf = lane >> 4;
  const int l16   = lane & 15;
  const long nTile0 = (long)(n0 >> 4);

  // This lane's two A rows and their embedding rows for timestep t.
  const int row0 = m0 + l16;
  const int row1 = row0 + 16;
  const int tk0 = (row0 < BATCH) ? qtok[row0 * SEQT + t]
                                 : rtok[(row0 - BATCH) * SEQT + t];
  const int tk1 = (row1 < BATCH) ? qtok[row1 * SEQT + t]
                                 : rtok[(row1 - BATCH) * SEQT + t];
  const float*  e0 = emb + (long)tk0 * EDIM + lhalf * 8;
  const float*  e1 = emb + (long)tk1 * EDIM + lhalf * 8;
  const bf16_t* h0 = H + (long)row0 * HDIM + lhalf * 8;
  const bf16_t* h1 = H + (long)row1 * HDIM + lhalf * 8;

  // Hoisted per-column B base pointers (per-kt step is a constant 1 KiB).
  const bf16_t* bbase[4];
#pragma unroll
  for (int j = 0; j < 4; ++j)
    bbase[j] = Bp + (nTile0 + j) * (long)kTiles * 512 + lane * 16;

  v8f acc[2][4];
#pragma unroll
  for (int i = 0; i < 2; ++i)
#pragma unroll
    for (int j = 0; j < 4; ++j)
#pragma unroll
      for (int v = 0; v < 8; ++v) acc[i][j][v] = 0.0f;

  auto loadA = [&](int kt, v16bf& a0, v16bf& a1) {
    if (kt < embKt) {                       // wave-uniform branch
      a0 = load_emb_frag(e0 + kt * 32);
      a1 = load_emb_frag(e1 + kt * 32);
    } else {
      int ko = (kt - embKt) * 32;
      a0 = load_afrag(h0 + ko);
      a1 = load_afrag(h1 + ko);
    }
  };
  auto loadB = [&](int kt, v16bf* b) {
#pragma unroll
    for (int j = 0; j < 4; ++j)
      b[j] = *(const v16bf*)(bbase[j] + (long)kt * 512);
  };
  auto mm8 = [&](const v16bf& a0, const v16bf& a1, const v16bf* b) {
#pragma unroll
    for (int j = 0; j < 4; ++j) {
      acc[0][j] = __builtin_amdgcn_wmma_f32_16x16x32_bf16(
          false, a0, false, b[j], (short)0, acc[0][j], false, false);
      acc[1][j] = __builtin_amdgcn_wmma_f32_16x16x32_bf16(
          false, a1, false, b[j], (short)0, acc[1][j], false, false);
    }
  };

  // --- ping-pong pipelined K loop (kTiles is even: 24 or 16) ---------------
  v16bf a0A, a1A, bA[4], a0B, a1B, bB[4];
  loadA(0, a0A, a1A); loadB(0, bA);
  loadA(1, a0B, a1B); loadB(1, bB);
  for (int kt = 0; kt < kTiles - 2; kt += 2) {
    mm8(a0A, a1A, bA);
    loadA(kt + 2, a0A, a1A); loadB(kt + 2, bA);
    mm8(a0B, a1B, bB);
    loadA(kt + 3, a0B, a1B); loadB(kt + 3, bB);
  }
  mm8(a0A, a1A, bA);
  mm8(a0B, a1B, bB);

#pragma unroll
  for (int i = 0; i < 2; ++i)
#pragma unroll
    for (int j = 0; j < 4; ++j)
#pragma unroll
      for (int v = 0; v < 8; ++v)
        Z[(long)(m0 + i * 16 + v + lhalf * 8) * ldc + n0 + j * 16 + l16] =
            acc[i][j][v];
}

// C = A @ B^T, both row-major bf16. B fragment = contiguous row read (the
// transpose makes the WMMA-B layout a natural 32-byte row segment).
__global__ __launch_bounds__(256)
void gemm_bf16_abT(const bf16_t* __restrict__ A, int lda,
                   const bf16_t* __restrict__ Bm, int ldb,
                   float* __restrict__ C, int ldc, int kTiles) {
  const int lane  = threadIdx.x & 31;
  const int wave  = threadIdx.x >> 5;
  const int m0    = blockIdx.x * 64 + (wave >> 2) * 32;
  const int n0    = blockIdx.y * 256 + (wave & 3) * 64;
  const int lhalf = lane >> 4;
  const int l16   = lane & 15;

  v8f acc[2][4];
#pragma unroll
  for (int i = 0; i < 2; ++i)
#pragma unroll
    for (int j = 0; j < 4; ++j)
#pragma unroll
      for (int v = 0; v < 8; ++v) acc[i][j][v] = 0.0f;

  const bf16_t* arow0 = A + (long)(m0 + l16) * lda + lhalf * 8;
  const bf16_t* arow1 = arow0 + (long)16 * lda;

  for (int kt = 0; kt < kTiles; ++kt) {
    v16bf a0 = load_afrag(arow0 + kt * 32);
    v16bf a1 = load_afrag(arow1 + kt * 32);
#pragma unroll
    for (int j = 0; j < 4; ++j) {
      const bf16_t* bp = Bm + (long)(n0 + j * 16 + l16) * ldb + kt * 32 + lhalf * 8;
      v16bf b = load_afrag(bp);
      acc[0][j] = __builtin_amdgcn_wmma_f32_16x16x32_bf16(
          false, a0, false, b, (short)0, acc[0][j], false, false);
      acc[1][j] = __builtin_amdgcn_wmma_f32_16x16x32_bf16(
          false, a1, false, b, (short)0, acc[1][j], false, false);
    }
  }
#pragma unroll
  for (int i = 0; i < 2; ++i)
#pragma unroll
    for (int j = 0; j < 4; ++j)
#pragma unroll
      for (int v = 0; v < 8; ++v)
        C[(long)(m0 + i * 16 + v + lhalf * 8) * ldc + n0 + j * 16 + l16] =
            acc[i][j][v];
}

// Peephole gates in f32; TF gate order i, j, f, o; freeze state at t >= len.
__global__ void lstm_gates(const float* __restrict__ Z, const float* __restrict__ bias,
                           const float* __restrict__ wi, const float* __restrict__ wf,
                           const float* __restrict__ wo,
                           const int* __restrict__ qlen, const int* __restrict__ rlen,
                           float* __restrict__ Cst, bf16_t* __restrict__ hbuf, int t) {
  int idx = blockIdx.x * blockDim.x + threadIdx.x;   // ROWS*HDIM
  int row = idx >> 9, hu = idx & (HDIM - 1);
  int b   = row & (BATCH - 1);
  int len = (row < BATCH ? qlen : rlen)[b];
  if (t >= len) return;                              // state frozen past length
  const float* zr = Z + (long)row * GDIM;
  float zi = zr[hu]            + bias[hu];
  float zj = zr[HDIM + hu]     + bias[HDIM + hu];
  float zf = zr[2 * HDIM + hu] + bias[2 * HDIM + hu];
  float zo = zr[3 * HDIM + hu] + bias[3 * HDIM + hu];
  float c  = Cst[idx];
  float ig = 1.0f / (1.0f + __expf(-(zi + wi[hu] * c)));
  float fg = 1.0f / (1.0f + __expf(-(zf + 2.0f + wf[hu] * c)));
  float cn = fg * c + ig * tanhf(zj);
  float og = 1.0f / (1.0f + __expf(-(zo + wo[hu] * cn)));
  float hn = og * tanhf(cn);
  Cst[idx]  = cn;
  hbuf[idx] = (bf16_t)hn;                            // h feeds next-step GEMM
}

__global__ void f32_to_bf16(const float* __restrict__ s, bf16_t* __restrict__ d,
                            int n) {
  int idx = blockIdx.x * blockDim.x + threadIdx.x;
  if (idx < n) d[idx] = (bf16_t)s[idx];
}

__global__ void take_diag(const float* __restrict__ D, float* __restrict__ out) {
  int i = threadIdx.x;
  out[i] = D[i * BATCH + i];
}

extern "C" void kernel_launch(void* const* d_in, const int* in_sizes, int n_in,
                              void* d_out, int out_size, void* d_ws, size_t ws_size,
                              hipStream_t stream) {
  const int*   qtok = (const int*)d_in[0];
  const int*   rtok = (const int*)d_in[1];
  const int*   qlen = (const int*)d_in[2];
  const int*   rlen = (const int*)d_in[3];
  const float* emb  = (const float*)d_in[4];
  const float* W    = (const float*)d_in[5];
  const float* bia  = (const float*)d_in[6];
  const float* wi   = (const float*)d_in[7];
  const float* wf   = (const float*)d_in[8];
  const float* wo   = (const float*)d_in[9];
  const float* Mm   = (const float*)d_in[10];
  float* out = (float*)d_out;

  char*  ws  = (char*)d_ws;
  size_t off = 0;
  auto alloc = [&](size_t bytes) -> void* {
    void* p = ws + off;
    off = (off + bytes + 255) & ~(size_t)255;
    return p;
  };
  bf16_t* Wpack = (bf16_t*)alloc((size_t)KDIM * GDIM * 2);   // 3 MB
  bf16_t* Mpack = (bf16_t*)alloc((size_t)HDIM * HDIM * 2);   // 0.5 MB
  bf16_t* Hbuf  = (bf16_t*)alloc((size_t)ROWS * HDIM * 2);   // h state, bf16
  float*  Zbuf  = (float*)alloc((size_t)ROWS * GDIM * 4);    // 4 MB
  float*  Cst   = (float*)alloc((size_t)ROWS * HDIM * 4);    // cell state, f32
  float*  Qproj = (float*)alloc((size_t)BATCH * HDIM * 4);
  bf16_t* Qpbf  = (bf16_t*)alloc((size_t)BATCH * HDIM * 2);

  repack_w<<<(KDIM * GDIM + 255) / 256, 256, 0, stream>>>(W, Wpack, KDIM, GDIM);
  repack_w<<<(HDIM * HDIM + 255) / 256, 256, 0, stream>>>(Mm, Mpack, HDIM, HDIM);
  init_state<<<(ROWS * HDIM) / 256, 256, 0, stream>>>(Cst, Hbuf);

  for (int t = 0; t < SEQT; ++t) {
    gemm_fused<<<dim3(ROWS / 64, GDIM / 256), 256, 0, stream>>>(
        qtok, rtok, emb, t, Hbuf, Wpack, Zbuf, GDIM, KDIM / 32, EDIM / 32);
    lstm_gates<<<(ROWS * HDIM) / 256, 256, 0, stream>>>(
        Zbuf, bia, wi, wf, wo, qlen, rlen, Cst, Hbuf, t);
  }

  // q_proj = q_enc @ M  (same GEMM kernel, no embedding phase: embKt = 0)
  gemm_fused<<<dim3(BATCH / 64, HDIM / 256), 256, 0, stream>>>(
      qtok, rtok, emb, 0, Hbuf, Mpack, Qproj, HDIM, HDIM / 32, 0);
  f32_to_bf16<<<(BATCH * HDIM) / 256, 256, 0, stream>>>(Qproj, Qpbf, BATCH * HDIM);

  // distances = q_proj @ r_enc^T ; echo = q_proj @ q_enc^T (e_enc == q_enc)
  gemm_bf16_abT<<<dim3(BATCH / 64, 1), 256, 0, stream>>>(
      Qpbf, HDIM, Hbuf + (size_t)BATCH * HDIM, HDIM, out, BATCH, HDIM / 32);
  gemm_bf16_abT<<<dim3(BATCH / 64, 1), 256, 0, stream>>>(
      Qpbf, HDIM, Hbuf, HDIM, out + BATCH * BATCH, BATCH, HDIM / 32);
  take_diag<<<1, 256, 0, stream>>>(out, out + 2 * BATCH * BATCH);
}
